// KoLeoLoss_61701500175092
// MI455X (gfx1250) — matrix operand order repaired and checked by reading, not compiled, not run on hardware
//
#include <hip/hip_runtime.h>
#include <hip/hip_bf16.h>
#include <math.h>

typedef __attribute__((ext_vector_type(16))) __bf16 v16bf;
typedef __attribute__((ext_vector_type(8)))  __bf16 v8bf;
typedef __attribute__((ext_vector_type(8)))  float  v8f;

#define KN 8192
#define KD 768
#define JSEG 4          // j-sweep split into 4 segments (blockIdx.y)
#define JSPAN (KN / JSEG)

// ---------------- Kernel 1: split fp32 -> bf16 hi/lo ----------------
__global__ void koleo_split(const float* __restrict__ x,
                            unsigned short* __restrict__ hi_raw,
                            unsigned short* __restrict__ lo_raw) {
  __bf16* hi = reinterpret_cast<__bf16*>(hi_raw);
  __bf16* lo = reinterpret_cast<__bf16*>(lo_raw);
  size_t idx = ((size_t)blockIdx.x * 256 + threadIdx.x) * 4;
  float4 v = *(const float4*)(x + idx);
  float f[4] = {v.x, v.y, v.z, v.w};
#pragma unroll
  for (int e = 0; e < 4; ++e) {
    __bf16 h = (__bf16)f[e];
    hi[idx + e] = h;
    lo[idx + e] = (__bf16)(f[e] - (float)h);
  }
}

// ---------------- Kernel 2: per-row sum of squares (fp32) ----------------
__global__ void koleo_rowsq(const float* __restrict__ x, float* __restrict__ sq) {
  __shared__ float s[256];
  const int row = blockIdx.x;
  float p = 0.f;
  for (int t = threadIdx.x; t < KD; t += 256) {
    float v = x[(size_t)row * KD + t];
    p += v * v;
  }
  s[threadIdx.x] = p;
  __syncthreads();
  for (int off = 128; off > 0; off >>= 1) {
    if (threadIdx.x < off) s[threadIdx.x] += s[threadIdx.x + off];
    __syncthreads();
  }
  if (threadIdx.x == 0) sq[row] = s[0];
}

// ---------------- Kernel 3: WMMA Gram sweep + running row-min ----------------
// Grid: (KN/64, JSEG), 256 threads (8 waves).
// Wave grid 2(m) x 4(n); each wave computes a 32x32 tile per j-iteration.
// dot = hi*hi + hi*lo + lo*hi  (split-bf16 ~fp32 accuracy, f32 accumulate)
__global__ __launch_bounds__(256)
void koleo_minsweep(const unsigned short* __restrict__ hi_raw,
                    const unsigned short* __restrict__ lo_raw,
                    const float* __restrict__ sq,
                    float* __restrict__ rowminP) {
  const __bf16* xhi = reinterpret_cast<const __bf16*>(hi_raw);
  const __bf16* xlo = reinterpret_cast<const __bf16*>(lo_raw);
  __shared__ float smin[64][4];

  const int tid   = threadIdx.x;
  const int wave  = tid >> 5;
  const int lane  = tid & 31;
  const int laneN = lane & 15;     // row (A) / col (B) within 16
  const int laneH = lane >> 4;     // K-half selector
  const int wm    = wave >> 2;     // 0..1 -> 32-row sub-strip
  const int wn    = wave & 3;      // 0..3 -> 32-col sub-strip
  const int blockRow = blockIdx.x * 64;
  const int rowBase  = blockRow + wm * 32;
  const int seg      = blockIdx.y;

  const v8f zc = {};
  float rmin[2][8];
#pragma unroll
  for (int mi = 0; mi < 2; ++mi)
#pragma unroll
    for (int r = 0; r < 8; ++r) rmin[mi][r] = INFINITY;

  const int jBeg = seg * JSPAN;
  const int jEnd = jBeg + JSPAN;

  for (int j0 = jBeg; j0 < jEnd; j0 += 128) {
    const int colBase = j0 + wn * 32;

    v8f c[2][2];
#pragma unroll
    for (int mi = 0; mi < 2; ++mi)
#pragma unroll
      for (int ni = 0; ni < 2; ++ni) c[mi][ni] = zc;

    for (int k0 = 0; k0 < KD; k0 += 32) {
      v16bf ah[2], al[2], bh[2], bl[2];
      // A fragments: 16x32 bf16 A in this code's contiguous k order (same as B, not ISA table order) — lane row = laneN,
      // K elements: base=laneH*16, elems 0..15 -> K base..base+15
#pragma unroll
      for (int mi = 0; mi < 2; ++mi) {
        size_t off = (size_t)(rowBase + mi * 16 + laneN) * KD + k0 + laneH * 16;
        union { v16bf v; v8bf h[2]; } uh, ul;
        uh.h[0] = *(const v8bf*)(xhi + off);
        uh.h[1] = *(const v8bf*)(xhi + off + 8);
        ul.h[0] = *(const v8bf*)(xlo + off);
        ul.h[1] = *(const v8bf*)(xlo + off + 8);
        ah[mi] = uh.v;
        al[mi] = ul.v;
      }
      // B fragments: 32x16 bf16 B in this code's contiguous k order (same as A) — lane col = laneN,
      // K contiguous: laneH*16 .. +15 (one 32-byte load)
#pragma unroll
      for (int ni = 0; ni < 2; ++ni) {
        size_t off = (size_t)(colBase + ni * 16 + laneN) * KD + k0 + laneH * 16;
        bh[ni] = *(const v16bf*)(xhi + off);
        bl[ni] = *(const v16bf*)(xlo + off);
      }
#pragma unroll
      for (int mi = 0; mi < 2; ++mi)
#pragma unroll
        for (int ni = 0; ni < 2; ++ni) {
          c[mi][ni] = __builtin_amdgcn_wmma_f32_16x16x32_bf16(
              false, ah[mi], false, bh[ni], (short)0, c[mi][ni], false, false);
          c[mi][ni] = __builtin_amdgcn_wmma_f32_16x16x32_bf16(
              false, ah[mi], false, bl[ni], (short)0, c[mi][ni], false, false);
          c[mi][ni] = __builtin_amdgcn_wmma_f32_16x16x32_bf16(
              false, al[mi], false, bh[ni], (short)0, c[mi][ni], false, false);
        }
    }

    // Fold this j-tile into the running per-row minimum of (sq[j] - 2*dot)
#pragma unroll
    for (int ni = 0; ni < 2; ++ni) {
      const int col = colBase + ni * 16 + laneN;
      const float sqj = sq[col];
#pragma unroll
      for (int mi = 0; mi < 2; ++mi) {
        const int row0 = rowBase + mi * 16 + laneH * 8;
#pragma unroll
        for (int r = 0; r < 8; ++r) {
          float cand = fmaf(-2.0f, c[mi][ni][r], sqj);
          cand = (col == row0 + r) ? INFINITY : cand;  // exclude self
          rmin[mi][r] = fminf(rmin[mi][r], cand);
        }
      }
    }
  }

  // Reduce across the 16 column-lanes (wave32: xor 1,2,4,8 stays within halves)
#pragma unroll
  for (int mi = 0; mi < 2; ++mi)
#pragma unroll
    for (int r = 0; r < 8; ++r) {
      float v = rmin[mi][r];
#pragma unroll
      for (int m = 1; m < 16; m <<= 1) v = fminf(v, __shfl_xor(v, m, 32));
      if (laneN == 0) smin[wm * 32 + mi * 16 + laneH * 8 + r][wn] = v;
    }
  __syncthreads();
  if (tid < 64) {
    float m = fminf(fminf(smin[tid][0], smin[tid][1]),
                    fminf(smin[tid][2], smin[tid][3]));
    rowminP[(size_t)seg * KN + blockRow + tid] = m;
  }
}

// ---------------- Kernel 4: final loss ----------------
__global__ void koleo_loss(const float* __restrict__ sq,
                           const float* __restrict__ rowminP,
                           float* __restrict__ out) {
  __shared__ float s[256];
  float acc = 0.f;
  for (int i = threadIdx.x; i < KN; i += 256) {
    float m = INFINITY;
#pragma unroll
    for (int sgi = 0; sgi < JSEG; ++sgi)
      m = fminf(m, rowminP[(size_t)sgi * KN + i]);
    float d2 = fmaxf(sq[i] + m, 0.f);
    acc += logf(sqrtf(d2) + 1e-8f);
  }
  s[threadIdx.x] = acc;
  __syncthreads();
  for (int off = 128; off > 0; off >>= 1) {
    if (threadIdx.x < off) s[threadIdx.x] += s[threadIdx.x + off];
    __syncthreads();
  }
  if (threadIdx.x == 0) out[0] = -s[0] / (float)KN;
}

extern "C" void kernel_launch(void* const* d_in, const int* in_sizes, int n_in,
                              void* d_out, int out_size, void* d_ws, size_t ws_size,
                              hipStream_t stream) {
  const float* x = (const float*)d_in[0];
  char* ws = (char*)d_ws;
  const size_t nElem = (size_t)KN * KD;

  // Workspace layout (~25.3 MB):
  unsigned short* xhi    = (unsigned short*)(ws);                    // nElem * 2 B
  unsigned short* xlo    = (unsigned short*)(ws + nElem * 2);        // nElem * 2 B
  float*          sq     = (float*)(ws + nElem * 4);                 // KN * 4 B
  float*          rowmin = (float*)(ws + nElem * 4 + (size_t)KN * 4);// JSEG * KN * 4 B
  float*          out    = (float*)d_out;

  koleo_split<<<(int)(nElem / 4 / 256), 256, 0, stream>>>(x, xhi, xlo);
  koleo_rowsq<<<KN, 256, 0, stream>>>(x, sq);
  dim3 grid(KN / 64, JSEG);
  koleo_minsweep<<<grid, 256, 0, stream>>>(xhi, xlo, sq, rowmin);
  koleo_loss<<<1, 256, 0, stream>>>(sq, rowmin, out);
}
